// ACTP_30167850287467
// MI455X (gfx1250) — matrix operand order repaired and checked by reading, not compiled, not run on hardware
//
#include <hip/hip_runtime.h>

// ============================================================================
// ACTP closed-loop LSTM rollout for MI455X (gfx1250, wave32, WMMA).
//
// Strategy: batch-parallel persistent workgroups. Each workgroup owns a
// 64-row batch slice and runs all 119 timesteps; recurrent state (c1,c2,
// h1,h2,out4) never leaves LDS. Weights are repacked once per call into
// f16 B-fragment layout (v_wmma_f32_16x16x32_f16 operand order, ISA
// 7.12.2/7.12.4: lane%16<->N, halves<->K), gate columns interleaved [i f g o]
// per 16-col chunk so the LSTM cell elementwise is fused on accumulators.
// Activations are stored in LDS in A-fragment layout so A operands are
// single contiguous ds loads. ~0.2 GB total HBM traffic -> on-chip bound.
// Scratch (d_ws) use: 736256 f16 packed weights + 1920 f32 biases (~1.5MB).
// ============================================================================

typedef __attribute__((ext_vector_type(16))) _Float16 v16h;
typedef __attribute__((ext_vector_type(8)))  float    v8f;

#define BTOT 4096
#define NSTEP 119
#define CTX 10
#define HD 200
#define FD 48
#define AD 6
#define BT 64            // batch rows per workgroup
#define MT 4             // 16-row tiles per workgroup
#define CSTR 212         // LDS row stride (floats) for c-state (conflict-dodged)
#define KBI 2            // K-blocks (of 32) for 48-wide inputs (pad->64)
#define KBH 7            // K-blocks for 200-wide inputs (pad->224)
#define JC 13            // 16-col gate chunks (200 -> 208)
#define NTG (JC*4)       // 52 gate tiles, interleaved [i f g o] per chunk
#define NT1 13           // fc1 output tiles (200 -> 208)
#define NT2 3            // fc2 output tiles (48)
#define NWAVES 8
#define NTHR 256
#define NWG (BTOT/BT)    // 64 workgroups

// packed-weight offsets in halves (tile = 32 lanes * 16 halves = 512)
#define OW1A 0            // gates1 <- inp   : 52*2*512
#define OW1B 53248        // gates1 <- h1    : 52*7*512
#define OW2A 239616       // gates2 <- h1    : 52*7*512
#define OW2B 425984       // gates2 <- tiled : 52*2*512
#define OW2C 479232       // gates2 <- h2prev: 52*7*512
#define OW3A 665600       // fc1 <- h2       : 13*7*512
#define OW3B 712192       // fc1 <- inp      : 13*2*512
#define OW4  725504       // fc2 <- out3     : 3*7*512
#define WP_HALVES 736256
#define OB1 0
#define OB2 832
#define OB3 1664
#define OB4 1872

#define FRAG_HALVES (2*MT*KBH*512 + 2*MT*KBH*512 + 2*MT*KBI*512 + MT*KBI*512 + MT*KBH*512)
#define SMEM_BYTES (2*BT*CSTR*4 + FRAG_HALVES*2)   // 276480 B < 320 KB/WGP

// ---------------------------------------------------------------------------
// fragment helpers
// ---------------------------------------------------------------------------
__device__ __forceinline__ v16h ldfrag(const _Float16* p, int tile, int lane) {
  return *(const v16h*)(p + (((tile << 5) + lane) << 4));
}
// scatter one f32 value into A-fragment layout (16-bit A 16x32, ISA 7.12.2):
// lane = m + 16*((j%32/8)&1), half = (j%8) + 8*((j%32)/16)
__device__ __forceinline__ void stfrag(_Float16* buf, int KB, int mt, int m, int j, float v) {
  int kb = j >> 5, jp = j & 31;
  int l = (m & 15) | (((jp >> 3) & 1) << 4);
  int h = (jp & 7) | ((jp >> 4) << 3);
  buf[((((mt * KB + kb) << 5) + l) << 4) + h] = (_Float16)v;
}
__device__ __forceinline__ v8f wmma16(v16h a, v16h b, v8f c) {
  return __builtin_amdgcn_wmma_f32_16x16x32_f16(false, a, false, b, (short)0, c, false, false);
}
__device__ __forceinline__ float sigf(float x) { return 1.0f / (1.0f + __expf(-x)); }
__device__ __forceinline__ float tanhfast(float x) {
  float e = __expf(-2.0f * x);
  return (1.0f - e) / (1.0f + e);
}

// ---------------------------------------------------------------------------
// weight / bias repack kernels (run once per call; write into d_ws)
// B-fragment layout per tile: [lane(lane%16 = N col)][half(+lane/16 = K within 32-block)]
// gate packs permute output cols into [i f g o] interleaved 16-col chunks
// ---------------------------------------------------------------------------
__global__ void pack_gate_w(const float* __restrict__ W, _Float16* __restrict__ dst,
                            int Kin, int kOff, int Kuse, int KB) {
  int idx = blockIdx.x * 256 + threadIdx.x;
  int tot = NTG * KB * 512;
  if (idx >= tot) return;
  int h = idx & 15;
  int l = (idx >> 4) & 31;
  int tkb = idx >> 9;
  int nt = tkb / KB;
  int kb = tkb - nt * KB;
  int g = nt & 3, jc = nt >> 2;
  int jcol = (jc << 4) + (l & 15);
  int k = (kb << 5) + ((h & 7) | ((l >> 4) << 3) | ((h >> 3) << 4));
  float v = 0.0f;
  if (jcol < HD && k < Kuse) v = W[(g * HD + jcol) * Kin + k + kOff];
  dst[idx] = (_Float16)v;
}

__global__ void pack_fc_w(const float* __restrict__ W, _Float16* __restrict__ dst,
                          int Kin, int kOff, int Kuse, int KB, int Nout, int NT) {
  int idx = blockIdx.x * 256 + threadIdx.x;
  int tot = NT * KB * 512;
  if (idx >= tot) return;
  int h = idx & 15;
  int l = (idx >> 4) & 31;
  int tkb = idx >> 9;
  int nt = tkb / KB;
  int kb = tkb - nt * KB;
  int n = (nt << 4) + (l & 15);
  int k = (kb << 5) + ((h & 7) | ((l >> 4) << 3) | ((h >> 3) << 4));
  float v = 0.0f;
  if (n < Nout && k < Kuse) v = W[n * Kin + k + kOff];
  dst[idx] = (_Float16)v;
}

__global__ void pack_bias_gate(const float* __restrict__ ba, const float* __restrict__ bb,
                               float* __restrict__ dst) {
  int idx = blockIdx.x * 256 + threadIdx.x;
  if (idx >= NTG * 16) return;
  int h = idx & 15;
  int nt = idx >> 4;
  int g = nt & 3, jc = nt >> 2;
  int jcol = (jc << 4) + h;
  dst[idx] = (jcol < HD) ? (ba[g * HD + jcol] + bb[g * HD + jcol]) : 0.0f;
}

__global__ void pack_bias_fc(const float* __restrict__ b, float* __restrict__ dst,
                             int Nout, int NT) {
  int idx = blockIdx.x * 256 + threadIdx.x;
  if (idx >= NT * 16) return;
  dst[idx] = (idx < Nout) ? b[idx] : 0.0f;
}

// ---------------------------------------------------------------------------
// persistent rollout kernel: one workgroup = one 64-row batch slice, all 119
// steps, 5 barriers/step, all state LDS-resident
// ---------------------------------------------------------------------------
extern "C" __global__ void __launch_bounds__(NTHR)
actp_rollout(const float* __restrict__ tact, const float* __restrict__ acts,
             const _Float16* __restrict__ wp, const float* __restrict__ bp,
             float* __restrict__ out) {
  extern __shared__ char smem_raw[];
  float* c1 = (float*)smem_raw;            // BT x CSTR
  float* c2 = c1 + BT * CSTR;
  _Float16* fb   = (_Float16*)(c2 + BT * CSTR);
  _Float16* h1f0 = fb;                               // 2 x MT*KBH*512
  _Float16* h2f0 = h1f0 + 2 * MT * KBH * 512;        // 2 x MT*KBH*512
  _Float16* inf0 = h2f0 + 2 * MT * KBH * 512;        // 2 x MT*KBI*512
  _Float16* tif  = inf0 + 2 * MT * KBI * 512;        // MT*KBI*512
  _Float16* o3f  = tif + MT * KBI * 512;             // MT*KBH*512

  const int tid  = threadIdx.x;
  const int lane = tid & 31;
  const int wv   = tid >> 5;
  const int ln16 = lane & 15;
  const int mrow = (lane >> 4) << 3;     // C/D tile: lanes 16-31 hold M+8
  const int wgb  = blockIdx.x * BT;

  const _Float16* W1A = wp + OW1A;  const _Float16* W1B = wp + OW1B;
  const _Float16* W2A = wp + OW2A;  const _Float16* W2B = wp + OW2B;
  const _Float16* W2C = wp + OW2C;  const _Float16* W3A = wp + OW3A;
  const _Float16* W3B = wp + OW3B;  const _Float16* W4  = wp + OW4;
  const float* B1 = bp + OB1;  const float* B2 = bp + OB2;
  const float* B3 = bp + OB3;  const float* B4 = bp + OB4;

  // zero-init state + fragment padding (h0=c0=0; pad cols stay 0 forever)
  for (int i = tid; i < 2 * BT * CSTR; i += NTHR) c1[i] = 0.0f;
  for (int i = tid; i < FRAG_HALVES; i += NTHR) fb[i] = (_Float16)0.0f;
  __syncthreads();

  for (int t = 0; t < NSTEP; ++t) {
    const int cur = t & 1;
    _Float16* infc = inf0 + cur * (MT * KBI * 512);
    _Float16* infn = inf0 + (cur ^ 1) * (MT * KBI * 512);
    _Float16* h1p  = h1f0 + (cur ^ 1) * (MT * KBH * 512);
    _Float16* h1c  = h1f0 + cur * (MT * KBH * 512);
    _Float16* h2p  = h2f0 + (cur ^ 1) * (MT * KBH * 512);
    _Float16* h2c  = h2f0 + cur * (MT * KBH * 512);

    // ---- stage inputs: tactiles (context frames) and tiled [act_t|state]x4
    if (t < CTX) {
      for (int e = tid; e < BT * FD; e += NTHR) {
        int r = e / FD, c = e - r * FD;
        float v = tact[((size_t)t * BTOT + (wgb + r)) * FD + c];
        stfrag(infc, KBI, r >> 4, r & 15, c, v);
      }
    }
    for (int e = tid; e < BT * FD; e += NTHR) {
      int r = e / FD, c = e - r * FD;
      int cm = c % 12;
      float v = (cm < AD)
          ? acts[((size_t)(t + 1) * BTOT + (wgb + r)) * AD + cm]
          : acts[(size_t)(wgb + r) * AD + (cm - AD)];
      stfrag(tif, KBI, r >> 4, r & 15, c, v);
    }
    __syncthreads();

    // ---- phase 1: gates1 = [inp|h1_prev]·W + b ; fused LSTM cell 1 ----
    for (int u = wv; u < MT * JC; u += NWAVES) {
      int mt = u & (MT - 1), jc = u >> 2;
      v8f acc[4];
#pragma unroll
      for (int g = 0; g < 4; ++g) {
        float bv = B1[(((jc << 2) + g) << 4) + ln16];
#pragma unroll
        for (int r = 0; r < 8; ++r) acc[g][r] = bv;
      }
#pragma unroll
      for (int kb = 0; kb < KBI + KBH; ++kb) {
        v16h a = (kb < KBI) ? ldfrag(infc, mt * KBI + kb, lane)
                            : ldfrag(h1p, mt * KBH + (kb - KBI), lane);
#pragma unroll
        for (int g = 0; g < 4; ++g) {
          int nt = (jc << 2) + g;
          v16h b = (kb < KBI) ? ldfrag(W1A, nt * KBI + kb, lane)
                              : ldfrag(W1B, nt * KBH + (kb - KBI), lane);
          acc[g] = wmma16(a, b, acc[g]);
        }
      }
      int jn = (jc << 4) + ln16;
      if (jn < HD) {
#pragma unroll
        for (int r = 0; r < 8; ++r) {
          int mm = r + mrow;
          int row = (mt << 4) + mm;
          float iv = sigf(acc[0][r]);
          float fv = sigf(acc[1][r]);
          float gv = tanhfast(acc[2][r]);
          float ov = sigf(acc[3][r]);
          float cn = fv * c1[row * CSTR + jn] + iv * gv;
          c1[row * CSTR + jn] = cn;
          stfrag(h1c, KBH, mt, mm, jn, ov * tanhfast(cn));
        }
      }
    }
    __syncthreads();

    // ---- phase 2: gates2 = [h1|tiled|h2_prev]·W + b ; fused LSTM cell 2 ----
    for (int u = wv; u < MT * JC; u += NWAVES) {
      int mt = u & (MT - 1), jc = u >> 2;
      v8f acc[4];
#pragma unroll
      for (int g = 0; g < 4; ++g) {
        float bv = B2[(((jc << 2) + g) << 4) + ln16];
#pragma unroll
        for (int r = 0; r < 8; ++r) acc[g][r] = bv;
      }
#pragma unroll
      for (int kb = 0; kb < KBH + KBI + KBH; ++kb) {
        v16h a = (kb < KBH) ? ldfrag(h1c, mt * KBH + kb, lane)
               : (kb < KBH + KBI) ? ldfrag(tif, mt * KBI + (kb - KBH), lane)
                                  : ldfrag(h2p, mt * KBH + (kb - KBH - KBI), lane);
#pragma unroll
        for (int g = 0; g < 4; ++g) {
          int nt = (jc << 2) + g;
          v16h b = (kb < KBH) ? ldfrag(W2A, nt * KBH + kb, lane)
                 : (kb < KBH + KBI) ? ldfrag(W2B, nt * KBI + (kb - KBH), lane)
                                    : ldfrag(W2C, nt * KBH + (kb - KBH - KBI), lane);
          acc[g] = wmma16(a, b, acc[g]);
        }
      }
      int jn = (jc << 4) + ln16;
      if (jn < HD) {
#pragma unroll
        for (int r = 0; r < 8; ++r) {
          int mm = r + mrow;
          int row = (mt << 4) + mm;
          float iv = sigf(acc[0][r]);
          float fv = sigf(acc[1][r]);
          float gv = tanhfast(acc[2][r]);
          float ov = sigf(acc[3][r]);
          float cn = fv * c2[row * CSTR + jn] + iv * gv;
          c2[row * CSTR + jn] = cn;
          stfrag(h2c, KBH, mt, mm, jn, ov * tanhfast(cn));
        }
      }
    }
    __syncthreads();

    // ---- phase 3: out3 = tanh([h2|inp]·fc1_w.T + b) ----
    for (int u = wv; u < MT * NT1; u += NWAVES) {
      int mt = u & (MT - 1), nt = u >> 2;
      v8f acc;
      {
        float bv = B3[(nt << 4) + ln16];
#pragma unroll
        for (int r = 0; r < 8; ++r) acc[r] = bv;
      }
#pragma unroll
      for (int kb = 0; kb < KBH + KBI; ++kb) {
        v16h a = (kb < KBH) ? ldfrag(h2c, mt * KBH + kb, lane)
                            : ldfrag(infc, mt * KBI + (kb - KBH), lane);
        v16h b = (kb < KBH) ? ldfrag(W3A, nt * KBH + kb, lane)
                            : ldfrag(W3B, nt * KBI + (kb - KBH), lane);
        acc = wmma16(a, b, acc);
      }
      int jn = (nt << 4) + ln16;   // < 208, pad cols produce tanh(0)=0
#pragma unroll
      for (int r = 0; r < 8; ++r)
        stfrag(o3f, KBH, mt, r + mrow, jn, tanhfast(acc[r]));
    }
    __syncthreads();

    // ---- phase 4: out4 = tanh(out3·fc2_w.T + b); emit + feed back ----
    for (int u = wv; u < MT * NT2; u += NWAVES) {
      int mt = u & (MT - 1), nt = u >> 2;
      v8f acc;
      {
        float bv = B4[(nt << 4) + ln16];
#pragma unroll
        for (int r = 0; r < 8; ++r) acc[r] = bv;
      }
#pragma unroll
      for (int kb = 0; kb < KBH; ++kb)
        acc = wmma16(ldfrag(o3f, mt * KBH + kb, lane),
                     ldfrag(W4, nt * KBH + kb, lane), acc);
      int jn = (nt << 4) + ln16;   // < 48
#pragma unroll
      for (int r = 0; r < 8; ++r) {
        int mm = r + mrow;
        float v = tanhfast(acc[r]);
        stfrag(infn, KBI, mt, mm, jn, v);   // next-step input (closed loop)
        if (t >= CTX - 1)
          out[((size_t)(t - (CTX - 1)) * BTOT + (wgb + (mt << 4) + mm)) * FD + jn] = v;
      }
    }
    __syncthreads();
  }
}

// ---------------------------------------------------------------------------
extern "C" void kernel_launch(void* const* d_in, const int* in_sizes, int n_in,
                              void* d_out, int out_size, void* d_ws, size_t ws_size,
                              hipStream_t stream) {
  (void)in_sizes; (void)n_in; (void)out_size; (void)ws_size;  // needs ~1.5 MB ws
  const float* tact = (const float*)d_in[0];
  const float* acts = (const float*)d_in[1];
  const float* Wih1 = (const float*)d_in[2];
  const float* Whh1 = (const float*)d_in[3];
  const float* bih1 = (const float*)d_in[4];
  const float* bhh1 = (const float*)d_in[5];
  const float* Wih2 = (const float*)d_in[6];
  const float* Whh2 = (const float*)d_in[7];
  const float* bih2 = (const float*)d_in[8];
  const float* bhh2 = (const float*)d_in[9];
  const float* fc1w = (const float*)d_in[10];
  const float* fc1b = (const float*)d_in[11];
  const float* fc2w = (const float*)d_in[12];
  const float* fc2b = (const float*)d_in[13];

  _Float16* wp = (_Float16*)d_ws;
  float* bp = (float*)((char*)d_ws + (size_t)WP_HALVES * sizeof(_Float16));

  auto nb = [](int n) { return (n + 255) / 256; };
  // gates1: inp (48) and h1 (200)
  pack_gate_w<<<nb(NTG*KBI*512), 256, 0, stream>>>(Wih1, wp + OW1A,  48,   0,  48, KBI);
  pack_gate_w<<<nb(NTG*KBH*512), 256, 0, stream>>>(Whh1, wp + OW1B, 200,   0, 200, KBH);
  // gates2: h1 part of W_ih2 (cols 0..199), tiled part (cols 200..247), h2 recurrent
  pack_gate_w<<<nb(NTG*KBH*512), 256, 0, stream>>>(Wih2, wp + OW2A, 248,   0, 200, KBH);
  pack_gate_w<<<nb(NTG*KBI*512), 256, 0, stream>>>(Wih2, wp + OW2B, 248, 200,  48, KBI);
  pack_gate_w<<<nb(NTG*KBH*512), 256, 0, stream>>>(Whh2, wp + OW2C, 200,   0, 200, KBH);
  // fc1: h2 part (cols 0..199) and inp part (cols 200..247); fc2
  pack_fc_w<<<nb(NT1*KBH*512), 256, 0, stream>>>(fc1w, wp + OW3A, 248,   0, 200, KBH, 200, NT1);
  pack_fc_w<<<nb(NT1*KBI*512), 256, 0, stream>>>(fc1w, wp + OW3B, 248, 200,  48, KBI, 200, NT1);
  pack_fc_w<<<nb(NT2*KBH*512), 256, 0, stream>>>(fc2w, wp + OW4,  200,   0, 200, KBH,  48, NT2);
  // biases (b_ih + b_hh fused; gate-permuted; zero-padded)
  pack_bias_gate<<<nb(NTG*16), 256, 0, stream>>>(bih1, bhh1, bp + OB1);
  pack_bias_gate<<<nb(NTG*16), 256, 0, stream>>>(bih2, bhh2, bp + OB2);
  pack_bias_fc<<<nb(NT1*16), 256, 0, stream>>>(fc1b, bp + OB3, 200, NT1);
  pack_bias_fc<<<nb(NT2*16), 256, 0, stream>>>(fc2b, bp + OB4, 48, NT2);

  // 276 KB dynamic LDS (< 320 KB/WGP on CDNA5); attribute set is capture-safe
  hipFuncSetAttribute((const void*)actp_rollout,
                      hipFuncAttributeMaxDynamicSharedMemorySize, SMEM_BYTES);
  actp_rollout<<<NWG, NTHR, SMEM_BYTES, stream>>>(tact, acts, wp, bp, (float*)d_out);
}
